// GCN_57982058496638
// MI455X (gfx1250) — matrix operand order repaired and checked by reading, not compiled, not run on hardware
//
#include <hip/hip_runtime.h>

// -------------------- problem constants --------------------
#define N_NODES 40000
#define N_EDGES 640000
#define D       128
#define N_GRAPHS 64
#define N_STRIPS (N_NODES / 16)   // 2500, exact

typedef _Float16 v16h __attribute__((ext_vector_type(16)));
typedef float    v8f  __attribute__((ext_vector_type(8)));
typedef float    v4f  __attribute__((ext_vector_type(4)));

// -------------------- degree / norm --------------------
__global__ void gcn_deg_init(float* __restrict__ deg) {
    int i = blockIdx.x * 256 + threadIdx.x;
    if (i < N_NODES) deg[i] = 1.0f;              // self-loop contributes 1
}

__global__ void gcn_deg_edges(const int* __restrict__ dst, float* __restrict__ deg) {
    int e = blockIdx.x * 256 + threadIdx.x;
    if (e < N_EDGES) atomicAdd(&deg[dst[e]], 1.0f);
}

__global__ void gcn_rsqrt(float* __restrict__ deg) {
    int i = blockIdx.x * 256 + threadIdx.x;
    if (i < N_NODES) deg[i] = rsqrtf(deg[i]);    // deg >= 1 always
}

// -------------------- generic zero (float4 granularity) --------------------
__global__ void gcn_zero4(v4f* __restrict__ p, int n4) {
    int i = blockIdx.x * 256 + threadIdx.x;
    if (i < n4) p[i] = (v4f){0.f, 0.f, 0.f, 0.f};
}

// -------------------- WMMA GEMM: out[N,128] = A[N,128] @ W[128,128] ------
// Markidis split in f16, f32 accumulate: A@W ~= Ah@Wh + Al@Wh + Ah@Wl.
// One wave computes a 16x128 row strip (8 N-tiles). W is staged into LDS
// pre-swizzled into the exact V_WMMA_F32_16X16X32_F16 B-fragment layout.
__global__ __launch_bounds__(256) void gcn_gemm_wmma(const float* __restrict__ A,
                                                     const float* __restrict__ W,
                                                     float* __restrict__ Cout) {
    // B-fragment order: [kk(4)][nt(8)][lane(32)][i(16)] halves
    __shared__ _Float16 sBhi[4 * 8 * 32 * 16];   // 32 KB
    __shared__ _Float16 sBlo[4 * 8 * 32 * 16];   // 32 KB
    const int tid = threadIdx.x;

    // Stage + split W (coalesced reads over n).
    for (int p = tid; p < D * D; p += 256) {
        const int k = p >> 7;              // K row   0..127
        const int n = p & 127;             // N col   0..127 (fastest -> coalesced)
        const float w = W[p];
        const _Float16 hi = (_Float16)w;
        const _Float16 lo = (_Float16)(w - (float)hi);
        const int kk     = k >> 5;         // 32-wide K step
        const int kr     = k & 31;
        const int laneHi = (kr >> 3) & 1;  // lanes 16-31 hold K 8..15,24..31 of the step
        const int i      = (kr & 7) | ((kr >> 4) << 3);
        const int nt     = n >> 4;
        const int lane   = laneHi * 16 + (n & 15);
        const int idx    = (((kk * 8 + nt) * 32) + lane) * 16 + i;
        sBhi[idx] = hi;
        sBlo[idx] = lo;
    }
    __syncthreads();

    const int strip = blockIdx.x * 8 + (tid >> 5);
    if (strip >= N_STRIPS) return;                  // wave-uniform: EXEC stays all-1s
    const int lane    = tid & 31;
    const int rowBase = strip * 16;
    const int m       = rowBase + (lane & 15);      // A-fragment row for this lane
    const int kb      = (lane & 16) ? 8 : 0;        // A-fragment K sub-offset

    v8f acc[8] = {};                                // 8 N-tiles of 16x16 f32

    for (int kk = 0; kk < 4; ++kk) {
        // ---- A fragment: halves 0..7 = K kb..kb+7, halves 8..15 = K kb+16..kb+23
        const float* ap = A + m * D + kk * 32 + kb;
        float av[16];
        *(v4f*)(av + 0)  = *(const v4f*)(ap + 0);
        *(v4f*)(av + 4)  = *(const v4f*)(ap + 4);
        *(v4f*)(av + 8)  = *(const v4f*)(ap + 16);
        *(v4f*)(av + 12) = *(const v4f*)(ap + 20);
        v16h ahi, alo;
#pragma unroll
        for (int i = 0; i < 16; ++i) {
            _Float16 h = (_Float16)av[i];
            ahi[i] = h;
            alo[i] = (_Float16)(av[i] - (float)h);
        }
#pragma unroll
        for (int nt = 0; nt < 8; ++nt) {
            const int base = (((kk * 8 + nt) * 32) + lane) * 16;
            v16h bhi = *(const v16h*)&sBhi[base];   // 32B-aligned LDS loads
            v16h blo = *(const v16h*)&sBlo[base];
            acc[nt] = __builtin_amdgcn_wmma_f32_16x16x32_f16(
                false, ahi, false, bhi, (short)0, acc[nt], false, false);
            acc[nt] = __builtin_amdgcn_wmma_f32_16x16x32_f16(
                false, alo, false, bhi, (short)0, acc[nt], false, false);
            acc[nt] = __builtin_amdgcn_wmma_f32_16x16x32_f16(
                false, ahi, false, blo, (short)0, acc[nt], false, false);
        }
    }

    // C/D layout: lane L, VGPR v -> row = v + (L>=16 ? 8 : 0), col = L&15
    const int rowOff = (lane >> 4) * 8;
    const int col    = lane & 15;
#pragma unroll
    for (int nt = 0; nt < 8; ++nt)
#pragma unroll
        for (int v = 0; v < 8; ++v)
            Cout[(rowBase + rowOff + v) * D + nt * 16 + col] = acc[nt][v];
}

// -------------------- edge scatter: out[dst] += dinv[src]*dinv[dst]*h[src] ----
// One wave per edge; lane q moves a float4 (32 lanes * 4 = 128 floats).
__global__ void gcn_scatter(const int* __restrict__ src, const int* __restrict__ dst,
                            const float* __restrict__ dinv, const float* __restrict__ h,
                            float* __restrict__ out) {
    const int gid = blockIdx.x * 256 + threadIdx.x;
    int e = gid >> 5;                  // wave-uniform (blockDim multiple of 32)
    if (e >= N_EDGES) return;
    e = __builtin_amdgcn_readfirstlane(e);          // force scalar loads below
    const int q = gid & 31;
    const int s = src[e];
    const int d = dst[e];
    const float w = dinv[s] * dinv[d];
    const v4f v = ((const v4f*)(h + (size_t)s * D))[q];
    float* o = out + (size_t)d * D + q * 4;
    atomicAdd(o + 0, w * v.x);
    atomicAdd(o + 1, w * v.y);
    atomicAdd(o + 2, w * v.z);
    atomicAdd(o + 3, w * v.w);
}

// -------------------- add self-loop term + bias (+ optional ReLU) ----------
__global__ void gcn_finalize(float* __restrict__ agg, const float* __restrict__ h,
                             const float* __restrict__ dinv, const float* __restrict__ bias,
                             int doRelu) {
    const int gid = blockIdx.x * 256 + threadIdx.x;   // per float4
    if (gid >= N_NODES * (D / 4)) return;
    const int node = gid / (D / 4);
    const int q    = gid % (D / 4);
    const float selfw = dinv[node] * dinv[node];
    v4f a  = ((v4f*)agg)[gid];
    v4f hv = ((const v4f*)h)[gid];
    v4f b  = ((const v4f*)bias)[q];
    v4f r;
    r.x = a.x + selfw * hv.x + b.x;
    r.y = a.y + selfw * hv.y + b.y;
    r.z = a.z + selfw * hv.z + b.z;
    r.w = a.w + selfw * hv.w + b.w;
    if (doRelu) {
        r.x = fmaxf(r.x, 0.f); r.y = fmaxf(r.y, 0.f);
        r.z = fmaxf(r.z, 0.f); r.w = fmaxf(r.w, 0.f);
    }
    ((v4f*)agg)[gid] = r;
}

// -------------------- global mean pool --------------------
__global__ void gcn_pool_zero(float* __restrict__ out, float* __restrict__ counts) {
    int i = blockIdx.x * 256 + threadIdx.x;
    if (i < N_GRAPHS * D) out[i] = 0.f;
    if (i < N_GRAPHS)     counts[i] = 0.f;
}

__global__ void gcn_pool_add(const float* __restrict__ x, const int* __restrict__ batch,
                             float* __restrict__ out, float* __restrict__ counts) {
    const int gid = blockIdx.x * 256 + threadIdx.x;
    int node = gid >> 5;
    if (node >= N_NODES) return;
    node = __builtin_amdgcn_readfirstlane(node);
    const int q = gid & 31;
    const int g = batch[node];
    const v4f v = ((const v4f*)(x + (size_t)node * D))[q];
    float* o = out + g * D + q * 4;
    atomicAdd(o + 0, v.x);
    atomicAdd(o + 1, v.y);
    atomicAdd(o + 2, v.z);
    atomicAdd(o + 3, v.w);
    if (q == 0) atomicAdd(&counts[g], 1.0f);
}

__global__ void gcn_pool_div(float* __restrict__ out, const float* __restrict__ counts) {
    int i = blockIdx.x * 256 + threadIdx.x;
    if (i < N_GRAPHS * D) out[i] /= fmaxf(counts[i >> 7], 1.0f);
}

// -------------------- host launch --------------------
extern "C" void kernel_launch(void* const* d_in, const int* in_sizes, int n_in,
                              void* d_out, int out_size, void* d_ws, size_t ws_size,
                              hipStream_t stream) {
    const float* x     = (const float*)d_in[0];
    const int*   edge  = (const int*)d_in[1];
    const int*   src   = edge;               // edge_index[0]: message sources
    const int*   dst   = edge + N_EDGES;     // edge_index[1]: aggregation targets
    const int*   batch = (const int*)d_in[2];
    const float* W1    = (const float*)d_in[3];
    const float* b1    = (const float*)d_in[4];
    const float* W2    = (const float*)d_in[5];
    const float* b2    = (const float*)d_in[6];
    float*       out   = (float*)d_out;

    // workspace: dinv[N] | bufA[N*D] | bufB[N*D] | counts[G]   (~41.1 MB)
    float* ws     = (float*)d_ws;
    float* dinv   = ws;
    float* bufA   = dinv + N_NODES;
    float* bufB   = bufA + (size_t)N_NODES * D;
    float* counts = bufB + (size_t)N_NODES * D;

    const int T = 256;
    const int gN   = (N_NODES + T - 1) / T;
    const int gE   = (N_EDGES + T - 1) / T;
    const int gND4 = (N_NODES * (D / 4) + T - 1) / T;
    const int gEw  = (N_EDGES * 32 + T - 1) / T;
    const int gNw  = (N_NODES * 32 + T - 1) / T;
    const int gGD  = (N_GRAPHS * D + T - 1) / T;
    const int gGemm = (N_STRIPS + 7) / 8;     // 8 waves (strips) per block

    // symmetric norm
    gcn_deg_init <<<gN, T, 0, stream>>>(dinv);
    gcn_deg_edges<<<gE, T, 0, stream>>>(dst, dinv);
    gcn_rsqrt    <<<gN, T, 0, stream>>>(dinv);

    // ---- layer 1: h = relu(Ahat @ (x @ W1) + b1)
    gcn_gemm_wmma<<<gGemm, T, 0, stream>>>(x, W1, bufA);
    gcn_zero4    <<<gND4, T, 0, stream>>>((v4f*)bufB, N_NODES * (D / 4));
    gcn_scatter  <<<gEw, T, 0, stream>>>(src, dst, dinv, bufA, bufB);
    gcn_finalize <<<gND4, T, 0, stream>>>(bufB, bufA, dinv, b1, 1);

    // ---- layer 2: h = Ahat @ (h @ W2) + b2   (no ReLU)
    gcn_gemm_wmma<<<gGemm, T, 0, stream>>>(bufB, W2, bufA);
    gcn_zero4    <<<gND4, T, 0, stream>>>((v4f*)bufB, N_NODES * (D / 4));
    gcn_scatter  <<<gEw, T, 0, stream>>>(src, dst, dinv, bufA, bufB);
    gcn_finalize <<<gND4, T, 0, stream>>>(bufB, bufA, dinv, b2, 0);

    // ---- global mean pool
    gcn_pool_zero<<<gGD, T, 0, stream>>>(out, counts);
    gcn_pool_add <<<gNw, T, 0, stream>>>(bufB, batch, out, counts);
    gcn_pool_div <<<gGD, T, 0, stream>>>(out, counts);
}
